// ProDualNet_16690242913060
// MI455X (gfx1250) — matrix operand order repaired and checked by reading, not compiled, not run on hardware
//
#include <hip/hip_runtime.h>
#include <hip/hip_bf16.h>

#define Bd   4
#define Nd   2048
#define Kd   30
#define Hd   128
#define FFd  512
#define BN   (Bd*Nd)
#define NE   (BN*Kd)

typedef __attribute__((ext_vector_type(16))) __bf16 v16bf;
typedef __attribute__((ext_vector_type(8)))  float  v8f;
typedef unsigned int   u32;
typedef unsigned short u16;
typedef int vi4 __attribute__((vector_size(16)));   // GCC vector: matches builtin's V4i

// ---- CDNA5 async global->LDS copy path (guarded; falls back to VGPR copy) ----
#if defined(__has_builtin)
#if __has_builtin(__builtin_amdgcn_global_load_async_to_lds_b128) && \
    __has_builtin(__builtin_amdgcn_global_load_async_to_lds_b32)
#define USE_ASYNC_LDS 1
#endif
#endif

#define GAS1V(p) ((__attribute__((address_space(1))) vi4*)(unsigned long long)(p))
#define GAS1I(p) ((__attribute__((address_space(1))) int*)(unsigned long long)(p))
#define LAS3V(p) ((__attribute__((address_space(3))) vi4*)(p))
#define LAS3I(p) ((__attribute__((address_space(3))) int*)(p))

__device__ __forceinline__ void wait_asynccnt0(){
#if defined(USE_ASYNC_LDS)
#if __has_builtin(__builtin_amdgcn_s_wait_asynccnt)
  __builtin_amdgcn_s_wait_asynccnt(0);
#else
  asm volatile("s_wait_asynccnt 0" ::: "memory");
#endif
#endif
}

// ---------------- helpers ----------------
__device__ __forceinline__ u16 f2bf(float f){
  u32 u = __float_as_uint(f);
  u32 r = u + 0x7FFFu + ((u >> 16) & 1u);   // round-to-nearest-even
  return (u16)(r >> 16);
}
__device__ __forceinline__ u32 pack2bf(float lo, float hi){
  return (u32)f2bf(lo) | ((u32)f2bf(hi) << 16);
}
__device__ __forceinline__ float gelu_exact(float x){
  return 0.5f * x * (1.0f + erff(x * 0.70710678118654752f));
}
struct F3 { float x,y,z; };
__device__ __forceinline__ F3 sub3(F3 a, F3 b){ return {a.x-b.x, a.y-b.y, a.z-b.z}; }
__device__ __forceinline__ F3 cross3(F3 a, F3 b){
  return {a.y*b.z-a.z*b.y, a.z*b.x-a.x*b.z, a.x*b.y-a.y*b.x};
}
__device__ __forceinline__ float dot3(F3 a, F3 b){ return a.x*b.x+a.y*b.y+a.z*b.z; }
__device__ __forceinline__ F3 norm3z(F3 a){       // normalize, NaN(zero-len) -> 0
  float n2 = dot3(a,a);
  if (n2 > 0.f){ float r = rsqrtf(n2); return {a.x*r,a.y*r,a.z*r}; }
  return {0.f,0.f,0.f};
}
__device__ __forceinline__ float sgnf(float x){ return (x>0.f)?1.f:((x<0.f)?-1.f:0.f); }

// ---- WMMA fragment loaders (CDNA5 16x16x32 bf16 layouts, CDNA5 ISA §7.12.2) ----
// A (16xK rows in a dword buffer, row stride = strideDw dwords, 2 bf16 per dword):
// lane: m = lane%16, half = lane/16; dwords [kt*16+half*4 .. +3] and [kt*16+8+half*4 .. +3]
__device__ __forceinline__ v16bf load_a(const u32* base, int strideDw, int lane, int kt){
  int mm = lane & 15, hh = lane >> 4;
  const u32* p = base + mm*strideDw + kt*16 + hh*4;
  uint4 lo = *(const uint4*)p;
  uint4 hi = *(const uint4*)(p + 8);
  union { u32 u[8]; v16bf v; } U;
  U.u[0]=lo.x; U.u[1]=lo.y; U.u[2]=lo.z; U.u[3]=lo.w;
  U.u[4]=hi.x; U.u[5]=hi.y; U.u[6]=hi.z; U.u[7]=hi.w;
  return U.v;
}
// B: weights pre-packed so each lane's 8 dwords are contiguous (32B): two b128 loads
__device__ __forceinline__ v16bf load_b(const u32* __restrict__ P, int NT, int kt, int nt, int lane){
  const u32* p = P + ((((long)kt*NT + nt)*32 + lane) << 3);
  uint4 lo = *(const uint4*)p;
  uint4 hi = *(const uint4*)(p + 4);
  union { u32 u[8]; v16bf v; } U;
  U.u[0]=lo.x; U.u[1]=lo.y; U.u[2]=lo.z; U.u[3]=lo.w;
  U.u[4]=hi.x; U.u[5]=hi.y; U.u[6]=hi.z; U.u[7]=hi.w;
  return U.v;
}

// ---------------- weight preparation ----------------
// W1eff (288x128): rows 0-127 = W1[0:128] (hV_i), 128-255 = W1[256:384] (hV_j),
// 256-271 = We @ W1[128:256] (fused edge-embed), 272-287 = 0 (K pad to 32)
__global__ void k_w1eff(const float* __restrict__ W1, const float* __restrict__ We,
                        float* __restrict__ W1eff){
  int idx = blockIdx.x*blockDim.x + threadIdx.x;
  if (idx >= 288*128) return;
  int r = idx >> 7, n = idx & 127;
  float v = 0.f;
  if (r < 128)      v = W1[r*128 + n];
  else if (r < 256) v = W1[(r+128)*128 + n];
  else if (r < 272){
    int rr = r - 256;
    float s = 0.f;
    for (int c = 0; c < 128; c++) s = fmaf(We[rr*128+c], W1[(128+c)*128+n], s);
    v = s;
  }
  W1eff[idx] = v;
}
__global__ void k_bias1(const float* __restrict__ W1, const float* __restrict__ W1b,
                        const float* __restrict__ Web, float* __restrict__ b1eff){
  int n = blockIdx.x*blockDim.x + threadIdx.x;
  if (n >= 128) return;
  float s = W1b[n];
  for (int c = 0; c < 128; c++) s = fmaf(Web[c], W1[(128+c)*128+n], s);
  b1eff[n] = s;
}
// permute KxN row-major f32 weight into per-lane WMMA-B bf16 dword order
__global__ void k_pack(const float* __restrict__ W, u32* __restrict__ outp,
                       int Ncols, int NT, int total){
  int d = blockIdx.x*blockDim.x + threadIdx.x;
  if (d >= total) return;
  int v    = d & 7;
  int lane = (d >> 3) & 31;
  int nt   = (d >> 8) % NT;
  int kt   = (d >> 8) / NT;
  int k0 = kt*32 + (lane >> 4)*8 + (v >> 2)*16 + 2*(v & 3);
  int n  = nt*16 + (lane & 15);
  outp[d] = pack2bf(W[(long)k0*Ncols + n], W[(long)(k0+1)*Ncols + n]);
}
__global__ void k_zero(float* p, long nfl){
  long i = (long)blockIdx.x*blockDim.x + threadIdx.x;
  if (i < nfl) p[i] = 0.f;
}

// ---------------- node featurization (dihedrals/angles/frames) ----------------
__global__ void k_featurize(const float* __restrict__ X, float* __restrict__ feat,
                            float* __restrict__ Q){
  int id = blockIdx.x*blockDim.x + threadIdx.x;
  if (id >= BN) return;
  int n = id & (Nd-1);
  const float* Xn = X + (long)id*12;         // [4 atoms][3]
  // p[s] = Xf[3n-1+s], s=0..5
  F3 p[6];
  if (n > 0){ const float* q = Xn - 12 + 6; p[0] = {q[0],q[1],q[2]}; }
  else p[0] = {0,0,0};
  #pragma unroll
  for (int a = 0; a < 3; a++){ const float* q = Xn + a*3; p[1+a] = {q[0],q[1],q[2]}; }
  bool hasNext = (n < Nd-1);
  if (hasNext){ const float* q = Xn + 12; p[4] = {q[0],q[1],q[2]}; p[5] = {q[3],q[4],q[5]}; }
  else { p[4] = {0,0,0}; p[5] = {0,0,0}; }
  F3 u[5];
  #pragma unroll
  for (int s = 0; s < 5; s++) u[s] = norm3z(sub3(p[s+1], p[s]));

  const float eps = 1e-7f;
  float Dv[3], Av[3];
  #pragma unroll
  for (int pp = 0; pp < 3; pp++){
    long t = 3L*n - 1 + pp;
    bool valid = (t >= 0) && (t <= 3L*Nd - 4);
    if (valid){
      F3 u0 = u[pp], u1 = u[pp+1], u2 = u[pp+2];
      F3 c0 = norm3z(cross3(u0,u1));
      F3 c1 = norm3z(cross3(u1,u2));
      float cd = fminf(fmaxf(dot3(c0,c1), -1.f+eps), 1.f-eps);
      F3 vv = norm3z(cross3(c0,c1));
      Dv[pp] = sgnf(-dot3(vv,u1)) * acosf(cd);
      float ca = fminf(fmaxf(dot3(u0,u1), -1.f+eps), 1.f-eps);
      Av[pp] = acosf(ca);
    } else { Dv[pp] = 0.f; Av[pp] = 0.f; }
  }
  float* f = feat + (long)id*21;
  #pragma unroll
  for (int pp = 0; pp < 3; pp++){
    f[pp]   = cosf(Dv[pp]); f[3+pp] = sinf(Dv[pp]);
    f[6+pp] = cosf(Av[pp]); f[9+pp] = sinf(Av[pp]);
  }
  // local frame Q = [b1; n0; b1 x n0]  (zero row-pad at n = N-1, as in reference)
  F3 b1v = {0,0,0}, n0v = {0,0,0}, q2v = {0,0,0};
  if (hasNext){
    b1v = norm3z(sub3(u[1], u[2]));
    n0v = norm3z(cross3(u[1], u[2]));
    q2v = cross3(b1v, n0v);
  }
  float* Qo = Q + (long)id*9;
  Qo[0]=b1v.x; Qo[1]=b1v.y; Qo[2]=b1v.z;
  Qo[3]=n0v.x; Qo[4]=n0v.y; Qo[5]=n0v.z;
  Qo[6]=q2v.x; Qo[7]=q2v.y; Qo[8]=q2v.z;
  // V_direct: atoms {0,2,3} relative to atom0, frame-projected, normalized
  F3 Xa = p[1];
  const int atoms[3] = {0,2,3};
  #pragma unroll
  for (int a = 0; a < 3; a++){
    const float* q = Xn + atoms[a]*3;
    F3 d = { q[0]-Xa.x, q[1]-Xa.y, q[2]-Xa.z };
    F3 t = norm3z(F3{ dot3(b1v,d), dot3(n0v,d), dot3(q2v,d) });
    f[12+a*3+0] = t.x; f[12+a*3+1] = t.y; f[12+a*3+2] = t.z;
  }
}

// ---------------- hV = feat(21) @ Wv + b  (f32 + bf16 copies) ----------------
__global__ void k_nodeembed(const float* __restrict__ feat, const float* __restrict__ Wv,
                            const float* __restrict__ bv, float* __restrict__ hV,
                            u16* __restrict__ hVb){
  long id = (long)blockIdx.x*blockDim.x + threadIdx.x;
  if (id >= (long)BN*128) return;
  long node = id >> 7; int h = (int)(id & 127);
  const float* f = feat + node*21;
  float acc = bv[h];
  #pragma unroll
  for (int r = 0; r < 21; r++) acc = fmaf(f[r], Wv[r*128+h], acc);
  hV[id] = acc; hVb[id] = f2bf(acc);
}

// ---------------- edge features: 12 dirs + quaternion, bf16, K-padded to 32 ----------------
__global__ void k_edgefeat(const float* __restrict__ X, const int* __restrict__ eidx,
                           const float* __restrict__ Q, u16* __restrict__ edq){
  int e = blockIdx.x*blockDim.x + threadIdx.x;
  if (e >= NE) return;
  int iflat = e / Kd;
  int bb = iflat >> 11;                       // N = 2048
  int jflat = (bb << 11) + eidx[e];
  const float* Qi = Q + (long)iflat*9;
  const float* Qj = Q + (long)jflat*9;
  const float* Xi = X + (long)iflat*12;
  const float* Xj = X + (long)jflat*12;
  F3 Xa = {Xi[0],Xi[1],Xi[2]};
  F3 r0 = {Qi[0],Qi[1],Qi[2]};
  F3 r1 = {Qi[3],Qi[4],Qi[5]};
  F3 r2 = {Qi[6],Qi[7],Qi[8]};
  float fv[16];
  const int order[4] = {1,0,2,3};             // CA, N, C, O of neighbor
  #pragma unroll
  for (int a = 0; a < 4; a++){
    const float* q = Xj + order[a]*3;
    F3 d = { q[0]-Xa.x, q[1]-Xa.y, q[2]-Xa.z };
    F3 t = norm3z(F3{ dot3(r0,d), dot3(r1,d), dot3(r2,d) });
    fv[a*3+0] = t.x; fv[a*3+1] = t.y; fv[a*3+2] = t.z;
  }
  // R = Qi^T * Qj
  float R[3][3];
  #pragma unroll
  for (int i = 0; i < 3; i++)
    #pragma unroll
    for (int l = 0; l < 3; l++)
      R[i][l] = Qi[0+i]*Qj[0+l] + Qi[3+i]*Qj[3+l] + Qi[6+i]*Qj[6+l];
  float Rxx=R[0][0], Ryy=R[1][1], Rzz=R[2][2];
  float m0 = 0.5f*sqrtf(fabsf(1.f+Rxx-Ryy-Rzz));
  float m1 = 0.5f*sqrtf(fabsf(1.f-Rxx+Ryy-Rzz));
  float m2 = 0.5f*sqrtf(fabsf(1.f-Rxx-Ryy+Rzz));
  float qx = sgnf(R[2][1]-R[1][2])*m0;
  float qy = sgnf(R[0][2]-R[2][0])*m1;
  float qz = sgnf(R[1][0]-R[0][1])*m2;
  float qw = 0.5f*sqrtf(fmaxf(0.f, 1.f+Rxx+Ryy+Rzz));
  float n2 = qx*qx+qy*qy+qz*qz+qw*qw;
  float rn = (n2 > 0.f) ? rsqrtf(n2) : 0.f;
  fv[12]=qx*rn; fv[13]=qy*rn; fv[14]=qz*rn; fv[15]=qw*rn;
  u32* o = (u32*)(edq + (long)e*32);
  #pragma unroll
  for (int t = 0; t < 8; t++) o[t] = pack2bf(fv[2*t], fv[2*t+1]);
  #pragma unroll
  for (int t = 8; t < 16; t++) o[t] = 0u;
}

// ---------------- edge message MLP: K=288 -> 128 -> 128 -> 128 (WMMA bf16) ----------------
__global__ void __launch_bounds__(256) k_msg(const u16* __restrict__ hVb, const u16* __restrict__ edq,
                      const int* __restrict__ eidx,
                      const u32* __restrict__ pW1, const u32* __restrict__ pW2,
                      const u32* __restrict__ pW3,
                      const float* __restrict__ b1, const float* __restrict__ b2,
                      const float* __restrict__ b3, float* __restrict__ dh){
  __shared__ u32 A1[16*148];   // 16 rows x 144 dwords (288 bf16) + bank pad
  __shared__ u32 A2[16*68];    // 16 rows x 64 dwords (128 bf16) + bank pad
  const int tid  = threadIdx.x;
  const int wave = tid >> 5;
  const int lane = tid & 31;
  const int e0   = blockIdx.x * 16;
  __builtin_prefetch(pW2, 0, 0);
  __builtin_prefetch(pW3, 0, 0);
  // ---- stage A tile: [hV_i(128) | hV_j(128) | EdQ(32)] bf16, 16 edges ----
  {
    int row = tid >> 4, sub = tid & 15;
    int e = e0 + row;
    int iflat = e / Kd;
    int bb2 = iflat >> 11;
    int jflat = (bb2 << 11) + eidx[e];
    const uint4* si = (const uint4*)(hVb + (long)iflat*128) + sub;
    const uint4* sj = (const uint4*)(hVb + (long)jflat*128) + sub;
    const u32*  se = (const u32*)(edq + (long)e*32) + sub;
    u32* di = &A1[row*148 + sub*4];
    u32* dj = &A1[row*148 + 64 + sub*4];
    u32* de = &A1[row*148 + 128 + sub];
#if defined(USE_ASYNC_LDS)
    // CDNA5 async copy: global -> LDS, per-lane addresses, tracked by ASYNCcnt
    __builtin_amdgcn_global_load_async_to_lds_b128(GAS1V(si), LAS3V(di), 0, 0);
    __builtin_amdgcn_global_load_async_to_lds_b128(GAS1V(sj), LAS3V(dj), 0, 0);
    __builtin_amdgcn_global_load_async_to_lds_b32 (GAS1I(se), LAS3I(de), 0, 0);
    wait_asynccnt0();
#else
    *(uint4*)di = *si;
    *(uint4*)dj = *sj;
    *de = *se;
#endif
  }
  __syncthreads();
  const int lm   = lane & 15;     // output column-in-tile
  const int half = lane >> 4;
  const int ncol = wave*16 + lm;  // output column 0..127 (wave owns 16 cols)
  // layer 1: 9 K-tiles of 32
  v8f c1 = {};
  #pragma unroll
  for (int kt = 0; kt < 9; kt++){
    v16bf a = load_a(A1, 148, lane, kt);
    v16bf w = load_b(pW1, 8, kt, wave, lane);
    c1 = __builtin_amdgcn_wmma_f32_16x16x32_bf16(false, a, false, w, (short)0, c1, false, false);
  }
  {
    u16* A2h = (u16*)A2;
    float bb = b1[ncol];
    #pragma unroll
    for (int r = 0; r < 8; r++){
      int mr = r + half*8;
      A2h[mr*136 + ncol] = f2bf(gelu_exact(c1[r] + bb));
    }
  }
  __syncthreads();
  // layer 2
  v8f c2 = {};
  #pragma unroll
  for (int kt = 0; kt < 4; kt++){
    v16bf a = load_a(A2, 68, lane, kt);
    v16bf w = load_b(pW2, 8, kt, wave, lane);
    c2 = __builtin_amdgcn_wmma_f32_16x16x32_bf16(false, a, false, w, (short)0, c2, false, false);
  }
  __syncthreads();
  {
    u16* A2h = (u16*)A2;
    float bb = b2[ncol];
    #pragma unroll
    for (int r = 0; r < 8; r++){
      int mr = r + half*8;
      A2h[mr*136 + ncol] = f2bf(gelu_exact(c2[r] + bb));
    }
  }
  __syncthreads();
  // layer 3 + mean-over-K accumulate
  v8f c3 = {};
  #pragma unroll
  for (int kt = 0; kt < 4; kt++){
    v16bf a = load_a(A2, 68, lane, kt);
    v16bf w = load_b(pW3, 8, kt, wave, lane);
    c3 = __builtin_amdgcn_wmma_f32_16x16x32_bf16(false, a, false, w, (short)0, c3, false, false);
  }
  {
    float bb = b3[ncol];
    const float inv = 1.0f / (float)Kd;
    #pragma unroll
    for (int r = 0; r < 8; r++){
      int mr = r + half*8;
      int e = e0 + mr;
      int node = e / Kd;
      unsafeAtomicAdd(&dh[(long)node*128 + ncol], (c3[r] + bb)*inv);
    }
  }
}

// ---------------- node update: LN1 -> FF(128->512->128) -> +res -> LN2 ----------------
__global__ void __launch_bounds__(256) k_nodeff(const float* __restrict__ hV, const float* __restrict__ dh,
                         const u32* __restrict__ pWi, const u32* __restrict__ pWo,
                         const float* __restrict__ bi, const float* __restrict__ bo,
                         const float* __restrict__ g1, const float* __restrict__ be1,
                         const float* __restrict__ g2, const float* __restrict__ be2,
                         float* __restrict__ out){
  __shared__ float XLN[16*128];
  __shared__ u32   XB[16*68];     // bf16 LN1 output
  __shared__ u32   Hh[16*260];    // bf16 hidden 16x512 + bank pad
  __shared__ float Y[16*128];
  __shared__ float st[32];
  int tid = threadIdx.x, wave = tid >> 5, lane = tid & 31;
  long node0 = (long)blockIdx.x * 16;
  {
    int row = tid >> 4, sub = tid & 15;
    long base = (node0+row)*128 + sub*8;
    #pragma unroll
    for (int t = 0; t < 8; t++) XLN[row*128 + sub*8 + t] = hV[base+t] + dh[base+t];
  }
  __syncthreads();
  if (tid < 16){
    float mu = 0.f;
    for (int c = 0; c < 128; c++) mu += XLN[tid*128+c];
    mu *= (1.f/128.f);
    float var = 0.f;
    for (int c = 0; c < 128; c++){ float d = XLN[tid*128+c]-mu; var += d*d; }
    var *= (1.f/128.f);
    st[tid*2] = mu; st[tid*2+1] = rsqrtf(var + 1e-5f);
  }
  __syncthreads();
  {
    int row = tid >> 4, sub = tid & 15;
    float mu = st[row*2], rs = st[row*2+1];
    #pragma unroll
    for (int t = 0; t < 4; t++){
      int c0 = sub*8 + t*2;
      float a = (XLN[row*128+c0  ]-mu)*rs*g1[c0  ]+be1[c0  ];
      float b = (XLN[row*128+c0+1]-mu)*rs*g1[c0+1]+be1[c0+1];
      XLN[row*128+c0]   = a;
      XLN[row*128+c0+1] = b;
      XB[row*68 + sub*4 + t] = pack2bf(a, b);
    }
  }
  __syncthreads();
  const int lm = lane & 15, half = lane >> 4;
  u16* Hp = (u16*)Hh;
  // FF1: 16x128 @ 128x512, each wave does 4 N-tiles
  #pragma unroll
  for (int s = 0; s < 4; s++){
    int nt = wave*4 + s;
    v8f c = {};
    #pragma unroll
    for (int kt = 0; kt < 4; kt++){
      v16bf a = load_a(XB, 68, lane, kt);
      v16bf w = load_b(pWi, 32, kt, nt, lane);
      c = __builtin_amdgcn_wmma_f32_16x16x32_bf16(false, a, false, w, (short)0, c, false, false);
    }
    int n = nt*16 + lm;
    float bb = bi[n];
    #pragma unroll
    for (int r = 0; r < 8; r++){
      int mr = r + half*8;
      Hp[mr*520 + n] = f2bf(gelu_exact(c[r] + bb));
    }
  }
  __syncthreads();
  // FF2: 16x512 @ 512x128 + residual
  {
    v8f c = {};
    #pragma unroll
    for (int kt = 0; kt < 16; kt++){
      v16bf a = load_a(Hh, 260, lane, kt);
      v16bf w = load_b(pWo, 8, kt, wave, lane);
      c = __builtin_amdgcn_wmma_f32_16x16x32_bf16(false, a, false, w, (short)0, c, false, false);
    }
    int n = wave*16 + lm;
    float bb = bo[n];
    #pragma unroll
    for (int r = 0; r < 8; r++){
      int mr = r + half*8;
      Y[mr*128 + n] = XLN[mr*128+n] + c[r] + bb;
    }
  }
  __syncthreads();
  if (tid < 16){
    float mu = 0.f;
    for (int c = 0; c < 128; c++) mu += Y[tid*128+c];
    mu *= (1.f/128.f);
    float var = 0.f;
    for (int c = 0; c < 128; c++){ float d = Y[tid*128+c]-mu; var += d*d; }
    var *= (1.f/128.f);
    st[tid*2] = mu; st[tid*2+1] = rsqrtf(var + 1e-5f);
  }
  __syncthreads();
  {
    int row = tid >> 4, sub = tid & 15;
    float mu = st[row*2], rs = st[row*2+1];
    long base = (node0+row)*128;
    #pragma unroll
    for (int t = 0; t < 8; t++){
      int c = sub*8 + t;
      out[base + c] = (Y[row*128+c]-mu)*rs*g2[c] + be2[c];
    }
  }
}

// ---------------- host launcher ----------------
extern "C" void kernel_launch(void* const* d_in, const int* in_sizes, int n_in,
                              void* d_out, int out_size, void* d_ws, size_t ws_size,
                              hipStream_t stream){
  const float* X    = (const float*)d_in[0];
  const int*   Eidx = (const int*)  d_in[1];
  const float* Wv_w = (const float*)d_in[2];
  const float* Wv_b = (const float*)d_in[3];
  const float* We_w = (const float*)d_in[4];
  const float* We_b = (const float*)d_in[5];
  const float* W1_w = (const float*)d_in[6];
  const float* W1_b = (const float*)d_in[7];
  const float* W2_w = (const float*)d_in[8];
  const float* W2_b = (const float*)d_in[9];
  const float* W3_w = (const float*)d_in[10];
  const float* W3_b = (const float*)d_in[11];
  const float* Wi_w = (const float*)d_in[12];
  const float* Wi_b = (const float*)d_in[13];
  const float* Wo_w = (const float*)d_in[14];
  const float* Wo_b = (const float*)d_in[15];
  const float* ln1g = (const float*)d_in[16];
  const float* ln1b = (const float*)d_in[17];
  const float* ln2g = (const float*)d_in[18];
  const float* ln2b = (const float*)d_in[19];
  float* out = (float*)d_out;
  (void)in_sizes; (void)n_in; (void)out_size; (void)ws_size;

  char* ws = (char*)d_ws;
  size_t off = 0;
  auto alloc = [&](size_t bytes)->char*{
    off = (off + 255) & ~(size_t)255;
    char* p = ws + off; off += bytes; return p;
  };
  float* Q     = (float*)alloc((size_t)BN*9*4);
  float* feat  = (float*)alloc((size_t)BN*21*4);
  float* hV    = (float*)alloc((size_t)BN*128*4);
  u16*   hVb   = (u16*)  alloc((size_t)BN*128*2);
  u16*   edq   = (u16*)  alloc((size_t)NE*32*2);
  float* dh    = (float*)alloc((size_t)BN*128*4);
  float* W1eff = (float*)alloc((size_t)288*128*4);
  float* b1eff = (float*)alloc((size_t)128*4);
  u32*   pW1   = (u32*)  alloc((size_t)9*8*256*4);
  u32*   pW2   = (u32*)  alloc((size_t)4*8*256*4);
  u32*   pW3   = (u32*)  alloc((size_t)4*8*256*4);
  u32*   pWi   = (u32*)  alloc((size_t)4*32*256*4);
  u32*   pWo   = (u32*)  alloc((size_t)16*8*256*4);

  // weight prep (re-run every call; deterministic, input-only)
  k_w1eff<<<(288*128+255)/256, 256, 0, stream>>>(W1_w, We_w, W1eff);
  k_bias1<<<1, 128, 0, stream>>>(W1_w, W1_b, We_b, b1eff);
  k_pack<<<(9*8*256)/256,  256, 0, stream>>>(W1eff, pW1, 128, 8,  9*8*256);
  k_pack<<<(4*8*256)/256,  256, 0, stream>>>(W2_w,  pW2, 128, 8,  4*8*256);
  k_pack<<<(4*8*256)/256,  256, 0, stream>>>(W3_w,  pW3, 128, 8,  4*8*256);
  k_pack<<<(4*32*256)/256, 256, 0, stream>>>(Wi_w,  pWi, 512, 32, 4*32*256);
  k_pack<<<(16*8*256)/256, 256, 0, stream>>>(Wo_w,  pWo, 128, 8,  16*8*256);
  k_zero<<<(BN*128)/256, 256, 0, stream>>>(dh, (long)BN*128);

  k_featurize<<<BN/256, 256, 0, stream>>>(X, feat, Q);
  k_nodeembed<<<(BN*128)/256, 256, 0, stream>>>(feat, Wv_w, Wv_b, hV, hVb);
  k_edgefeat<<<NE/256, 256, 0, stream>>>(X, Eidx, Q, edq);
  k_msg<<<NE/16, 256, 0, stream>>>(hVb, edq, Eidx, pW1, pW2, pW3, b1eff, W2_b, W3_b, dh);
  k_nodeff<<<BN/16, 256, 0, stream>>>(hV, dh, pWi, pWo, Wi_b, Wo_b,
                                      ln1g, ln1b, ln2g, ln2b, out);
}
